// ValueNetwork_69398081568820
// MI455X (gfx1250) — matrix operand order repaired and checked by reading, not compiled, not run on hardware
//
#include <hip/hip_runtime.h>
#include <hip/hip_bf16.h>
#include <math.h>

// ---------------------------------------------------------------------------
// MI455X (gfx1250) fused value-network forward.
// All GEMMs run on v_wmma_f32_16x16x32_bf16 (bf16 in, f32 accumulate).
// ---------------------------------------------------------------------------

typedef __attribute__((ext_vector_type(16))) __bf16 bf16x16;
typedef __attribute__((ext_vector_type(8)))  float  f32x8;

#define TPB    256
#define NWAVES 8
#define MAXT   5   // max WMMA tiles per wave in any layer (mlp1 L1: 40 tiles / 8 waves)

// ---- A fragment (16x32 bf16). ISA 7.12.2: lane L holds row M=L&15;
// element e covers K = (e&7) + 8*(L>=16) + 16*(e>=8). stride==0 => broadcast row.
__device__ __forceinline__ bf16x16 load_A_frag(const __bf16* A, int stride,
                                               int rowbase, int k0, int lane) {
  int row = rowbase + (lane & 15);
  int hi  = (lane >> 4) << 3;
  const __bf16* p = A + row * stride + k0;
  bf16x16 a;
#pragma unroll
  for (int e = 0; e < 16; ++e)
    a[e] = p[(e & 7) + hi + ((e >> 3) << 4)];
  return a;
}

// ---- B fragment (32x16 bf16) from a transposed LDS stage wT[n][32]:
// lane L holds column N=L&15; element e covers K = (e&7) + 8*(L>=16) + 16*(e>=8).
__device__ __forceinline__ bf16x16 load_B_frag(const __bf16* wst, int n0, int lane) {
  const __bf16* p = wst + ((n0 + (lane & 15)) << 5) + ((lane >> 4) << 3);
  bf16x16 b;
#pragma unroll
  for (int e = 0; e < 16; ++e) b[e] = p[(e & 7) + ((e >> 3) << 4)];
  return b;
}

// ---- C/D (16x16 f32): lane L holds col N=L&15, VGPR r holds row M=r+8*(L>=16).
__device__ __forceinline__ void store_C_frag(const f32x8& c, __bf16* Out, int npad,
                                             int mt, int nt, const float* bias,
                                             int nout, bool relu, int lane) {
  int col = (nt << 4) + (lane & 15);
  int rb  = (mt << 4) + ((lane >> 4) << 3);
  float bv = (col < nout) ? bias[col] : 0.0f;
#pragma unroll
  for (int r = 0; r < 8; ++r) {
    float v = c[r] + bv;
    if (col >= nout) v = 0.0f;          // keep padded columns exactly zero
    if (relu) v = fmaxf(v, 0.0f);
    Out[(rb + r) * npad + col] = (__bf16)v;
  }
}

// ---- Block-cooperative GEMM: Out(M x NPAD) = relu(A(M x K) * W(kin x nout) + b)
// A is bf16 in LDS (row stride aStride, zero padded). W is f32 in global (L2-hot),
// staged transposed per 32-row K chunk into LDS. 8 waves split the tile grid.
// Compile-time tiling => scalar (SGPR) tile predicates, EXEC untouched at WMMA.
template <int MTILES, int KSTEPS, int NPAD>
__device__ __forceinline__ void gemm_tiles(
    const __bf16* A, int aStride, int kin,
    const float* W, int wrowoff, int nout, const float* bias,
    __bf16* Out, bool relu,
    __bf16* wstage, f32x8* acc, bool accInit, bool accStore, int tid)
{
  constexpr int NTN  = NPAD / 16;
  constexpr int NTOT = MTILES * NTN;
  constexpr int TPW  = (NTOT + NWAVES - 1) / NWAVES;  // <= MAXT
  static_assert(TPW <= MAXT, "accumulator array too small");

  const int lane = tid & 31;
  // Force the wave id into an SGPR: tile predicates become scalar branches and
  // EXEC is provably all-ones around every WMMA (ISA 7.12 requirement).
  const int wave = __builtin_amdgcn_readfirstlane(tid >> 5);

  if (accInit) {
#pragma unroll
    for (int t = 0; t < TPW; ++t)
#pragma unroll
      for (int r = 0; r < 8; ++r) acc[t][r] = 0.0f;
  }

  for (int ks = 0; ks < KSTEPS; ++ks) {
    __syncthreads();                       // previous consumers of wstage done
    constexpr int TOT = NPAD * 32;
    const int kk0 = ks * 32;
    for (int e = tid; e < TOT; e += TPB) { // coalesced global reads along n
      int k  = e / NPAD;                   // division by compile-time constant
      int n  = e - k * NPAD;
      int kk = kk0 + k;
      float v = 0.0f;
      if (kk < kin && n < nout) v = W[(size_t)(wrowoff + kk) * nout + n];
      wstage[(n << 5) + k] = (__bf16)v;    // transposed stage
    }
    __syncthreads();
#pragma unroll
    for (int t = 0; t < TPW; ++t) {
      const int tile = wave + t * NWAVES;  // scalar
      if (tile < NTOT) {                   // scalar predicate -> s_cbranch
        const int mt = tile / NTN;
        const int nt = tile - mt * NTN;
        bf16x16 af = load_A_frag(A, aStride, mt << 4, kk0, lane);
        bf16x16 bf = load_B_frag(wstage, nt << 4, lane);
        acc[t] = __builtin_amdgcn_wmma_f32_16x16x32_bf16(
            false, af, false, bf, (short)0, acc[t], false, false);
      }
    }
  }

  if (accStore) {
    __syncthreads();
#pragma unroll
    for (int t = 0; t < TPW; ++t) {
      const int tile = wave + t * NWAVES;
      if (tile < NTOT) {
        const int mt = tile / NTN;
        const int nt = tile - mt * NTN;
        store_C_frag(acc[t], Out, NPAD, mt, nt, bias, nout, relu, lane);
      }
    }
    __syncthreads();
  }
}

// ---------------------------------------------------------------------------
// Kernel 1: one block per batch (4096). local_map + mlp1 + attention + mlp2 +
// softmax pooling; writes joint[b][64] = [self_state(6), weighted(50), 0 pad].
// ---------------------------------------------------------------------------
struct P1 {
  const float *state;
  const float *W1a, *b1a, *W1b, *b1b;                       // mlp1
  const float *Wat1, *bat1, *Wat2, *bat2, *Wat3, *bat3;     // attention
  const float *W2a, *b2a, *W2b, *b2b;                       // mlp2
  float *joint;                                             // [4096][64]
};

__global__ __launch_bounds__(TPB) void vn_agents_kernel(P1 p) {
  __shared__ __bf16 shA[64 * 160];   // h1 -> m2h -> hat2
  __shared__ __bf16 shB[64 * 128];   // m1 -> m2
  __shared__ __bf16 shD[64 * 128];   // s (64x32) -> hat1
  __shared__ __bf16 wst[160 * 32];   // weight K-chunk stage (transposed)
  __shared__ __bf16 shg[128];        // g = mean(m1), zero padded
  __shared__ float  sxy[128];
  __shared__ float  se[64];
  __shared__ float  sw[64];
  __shared__ float  ssum;

  const int tid = threadIdx.x;
  const int b   = blockIdx.x;
  const float* st = p.state + (size_t)b * 64 * 13;

  // agent xy + raw state into s (bf16, stride 32, zero padded)
  if (tid < 128) sxy[tid] = st[(tid >> 1) * 13 + 6 + (tid & 1)];
  for (int e = tid; e < 64 * 32; e += TPB) {
    int i = e >> 5, k = e & 31;
    float v = (k < 13) ? st[i * 13 + k] : 0.0f;
    shD[e] = (__bf16)v;
  }
  __syncthreads();

  // local_map quadrant counts (dist<3, not self): cols 13..16 of s
  if (tid < 64) {
    int i = tid;
    float xi = sxy[2 * i], yi = sxy[2 * i + 1];
    float c0 = 0.f, c1 = 0.f, c2 = 0.f, c3 = 0.f;
    for (int j = 0; j < 64; ++j) {
      if (j == i) continue;
      float rx = sxy[2 * j] - xi, ry = sxy[2 * j + 1] - yi;
      if (rx * rx + ry * ry < 9.0f) {
        if (rx <= 0.f) { if (ry <= 0.f) c3 += 1.f; else c2 += 1.f; }
        else           { if (ry <= 0.f) c1 += 1.f; else c0 += 1.f; }
      }
    }
    shD[i * 32 + 13] = (__bf16)c0; shD[i * 32 + 14] = (__bf16)c1;
    shD[i * 32 + 15] = (__bf16)c2; shD[i * 32 + 16] = (__bf16)c3;
  }

  f32x8 acc[MAXT];
  // mlp1: (64x17)->150 relu ->100 relu (last_relu=True)
  gemm_tiles<4, 1, 160>(shD, 32,  17,  p.W1a, 0, 150, p.b1a, shA, true, wst, acc, true, true, tid);
  gemm_tiles<4, 5, 128>(shA, 160, 150, p.W1b, 0, 100, p.b1b, shB, true, wst, acc, true, true, tid);

  // g = mean over 64 agents of m1
  if (tid < 128) {
    float s = 0.f;
    for (int r = 0; r < 64; ++r) s += (float)shB[r * 128 + tid];
    shg[tid] = (__bf16)(s * (1.0f / 64.0f));
  }

  // attention L1: concat([m1, g]) @ Wat1(200x100): two K phases, same accumulators
  gemm_tiles<4, 4, 128>(shB, 128, 100, p.Wat1, 0,   100, p.bat1, shD, true, wst, acc, true,  false, tid);
  gemm_tiles<4, 4, 128>(shg, 0,   100, p.Wat1, 100, 100, p.bat1, shD, true, wst, acc, false, true,  tid);

  // mlp2: m1 ->100 relu ->50 (no relu)
  gemm_tiles<4, 4, 128>(shB, 128, 100, p.W2a, 0, 100, p.b2a, shA, true,  wst, acc, true, true, tid);
  gemm_tiles<4, 4, 64>(shA, 128, 100, p.W2b, 0, 50,  p.b2b, shB, false, wst, acc, true, true, tid);

  // attention L2: hat1 ->100 relu
  gemm_tiles<4, 4, 128>(shD, 128, 100, p.Wat2, 0, 100, p.bat2, shA, true, wst, acc, true, true, tid);

  // scores + masked softmax weights
  if (tid < 64) {
    float s = p.bat3[0];
    for (int k = 0; k < 100; ++k) s += (float)shA[tid * 128 + k] * p.Wat3[k];
    se[tid] = (s != 0.0f) ? expf(s) : 0.0f;
  }
  __syncthreads();
  if (tid == 0) {
    float s = 0.f;
    for (int i = 0; i < 64; ++i) s += se[i];
    ssum = s;
  }
  __syncthreads();
  if (tid < 64) sw[tid] = se[tid] / ssum;
  __syncthreads();

  // weighted pooling of m2 + self-state -> joint row (stride 64, zero padded)
  float* jrow = p.joint + (size_t)b * 64;
  if (tid < 64) {
    int c = tid;
    if (c < 50) {
      float s = 0.f;
      for (int i = 0; i < 64; ++i) s += sw[i] * (float)shB[i * 64 + c];
      jrow[6 + c] = s;
    } else if (c < 58) {
      jrow[6 + c] = 0.0f;               // cols 56..63
    }
  }
  if (tid >= 64 && tid < 70) jrow[tid - 64] = st[tid - 64];  // self_state = s[:,0,:6]
}

// ---------------------------------------------------------------------------
// Kernel 2: batch head. 32 rows/block, 128 blocks: mlp3 -> value, mlp3a +
// adv_stream -> advantage.
// ---------------------------------------------------------------------------
struct P2 {
  const float *joint;
  const float *W31, *b31, *W32, *b32, *W33, *b33, *W34, *b34;   // mlp3
  const float *Wa1, *ba1, *Wa2, *ba2, *Wa3, *ba3;               // mlp3a
  const float *Wv1, *bv1, *Wv2, *bv2;                           // adv_stream
  float *value, *adv;
};

__global__ __launch_bounds__(TPB) void vn_head_kernel(P2 p) {
  __shared__ __bf16 jA[32 * 64];
  __shared__ __bf16 shA[32 * 160];
  __shared__ __bf16 shB[32 * 128];
  __shared__ __bf16 shD[32 * 128];
  __shared__ __bf16 wst[160 * 32];

  const int tid = threadIdx.x;
  const int r0  = blockIdx.x * 32;

  for (int e = tid; e < 32 * 64; e += TPB)
    jA[e] = (__bf16)p.joint[(size_t)r0 * 64 + e];

  f32x8 acc[MAXT];
  // mlp3: 56->150 relu ->100 relu ->100 relu ->1
  gemm_tiles<2, 2, 160>(jA,  64,  56,  p.W31, 0, 150, p.b31, shA, true, wst, acc, true, true, tid);
  gemm_tiles<2, 5, 128>(shA, 160, 150, p.W32, 0, 100, p.b32, shB, true, wst, acc, true, true, tid);
  gemm_tiles<2, 4, 128>(shB, 128, 100, p.W33, 0, 100, p.b33, shD, true, wst, acc, true, true, tid);
  if (tid < 32) {
    float v = p.b34[0];
    for (int k = 0; k < 100; ++k) v += (float)shD[tid * 128 + k] * p.W34[k];
    p.value[r0 + tid] = v;
  }
  // mlp3a: 56->150 relu ->100 relu ->80
  gemm_tiles<2, 2, 160>(jA,  64,  56,  p.Wa1, 0, 150, p.ba1, shA, true,  wst, acc, true, true, tid);
  gemm_tiles<2, 5, 128>(shA, 160, 150, p.Wa2, 0, 100, p.ba2, shB, true,  wst, acc, true, true, tid);
  gemm_tiles<2, 4, 96>(shB, 128, 100, p.Wa3, 0, 80,  p.ba3, shD, false, wst, acc, true, true, tid);
  // adv_stream: 80->80 relu ->1
  gemm_tiles<2, 3, 96>(shD, 96, 80, p.Wv1, 0, 80, p.bv1, shA, true, wst, acc, true, true, tid);
  if (tid < 32) {
    float a = p.bv2[0];
    for (int k = 0; k < 80; ++k) a += (float)shA[tid * 96 + k] * p.Wv2[k];
    p.adv[r0 + tid] = a;
  }
}

// ---------------------------------------------------------------------------
// Kernel 3: out = value + (adv - mean(adv)), mean over all 4096 entries.
// ---------------------------------------------------------------------------
__global__ __launch_bounds__(TPB) void vn_final_kernel(const float* __restrict__ value,
                                                       const float* __restrict__ adv,
                                                       float* __restrict__ out) {
  __shared__ float part[TPB];
  __shared__ float smean;
  const int tid = threadIdx.x;
  float s = 0.f;
  for (int i = tid; i < 4096; i += TPB) s += adv[i];
  part[tid] = s;
  __syncthreads();
  if (tid == 0) {
    float t = 0.f;
    for (int j = 0; j < TPB; ++j) t += part[j];
    smean = t / 4096.0f;
  }
  __syncthreads();
  float m = smean;
  for (int i = tid; i < 4096; i += TPB) out[i] = value[i] + adv[i] - m;
}

// ---------------------------------------------------------------------------
extern "C" void kernel_launch(void* const* d_in, const int* in_sizes, int n_in,
                              void* d_out, int out_size, void* d_ws, size_t ws_size,
                              hipStream_t stream) {
  (void)out_size; (void)ws_size;
  // Locate 'state' by its unique size; the remaining 32 arrays are the params.
  int si = 0;
  for (int i = 0; i < n_in; ++i)
    if (in_sizes[i] == 4096 * 64 * 13) { si = i; break; }

  const float* P[40];
  int psz[40];
  int np = 0;
  for (int i = 0; i < n_in && np < 40; ++i)
    if (i != si) { P[np] = (const float*)d_in[i]; psz[np] = in_sizes[i]; ++np; }

  // Two plausible pytree orders: alphabetical dict keys (jax tree_flatten) vs
  // insertion order. adv_stream's first array is 80*80=6400; mlp1's is 17*150=2550.
  int A_adv, A_att, A_m1, A_m2, A_m3, A_m3a;
  if (np > 0 && psz[0] == 80 * 80) {       // alphabetical: adv_stream first
    A_adv = 0; A_att = 4; A_m1 = 10; A_m2 = 14; A_m3 = 18; A_m3a = 26;
  } else {                                  // insertion: mlp1 first
    A_m1 = 0; A_m2 = 4; A_att = 8; A_m3 = 14; A_m3a = 22; A_adv = 28;
  }

  float* wsf   = (float*)d_ws;
  float* joint = wsf;                       // 4096*64 floats
  float* value = wsf + 4096 * 64;           // 4096 floats
  float* adv   = wsf + 4096 * 64 + 4096;    // 4096 floats

  P1 p1;
  p1.state = (const float*)d_in[si];
  p1.W1a = P[A_m1];     p1.b1a = P[A_m1 + 1];
  p1.W1b = P[A_m1 + 2]; p1.b1b = P[A_m1 + 3];
  p1.Wat1 = P[A_att];     p1.bat1 = P[A_att + 1];
  p1.Wat2 = P[A_att + 2]; p1.bat2 = P[A_att + 3];
  p1.Wat3 = P[A_att + 4]; p1.bat3 = P[A_att + 5];
  p1.W2a = P[A_m2];     p1.b2a = P[A_m2 + 1];
  p1.W2b = P[A_m2 + 2]; p1.b2b = P[A_m2 + 3];
  p1.joint = joint;

  P2 p2;
  p2.joint = joint;
  p2.W31 = P[A_m3];     p2.b31 = P[A_m3 + 1];
  p2.W32 = P[A_m3 + 2]; p2.b32 = P[A_m3 + 3];
  p2.W33 = P[A_m3 + 4]; p2.b33 = P[A_m3 + 5];
  p2.W34 = P[A_m3 + 6]; p2.b34 = P[A_m3 + 7];
  p2.Wa1 = P[A_m3a];     p2.ba1 = P[A_m3a + 1];
  p2.Wa2 = P[A_m3a + 2]; p2.ba2 = P[A_m3a + 3];
  p2.Wa3 = P[A_m3a + 4]; p2.ba3 = P[A_m3a + 5];
  p2.Wv1 = P[A_adv];     p2.bv1 = P[A_adv + 1];
  p2.Wv2 = P[A_adv + 2]; p2.bv2 = P[A_adv + 3];
  p2.value = value;
  p2.adv   = adv;

  vn_agents_kernel<<<4096, TPB, 0, stream>>>(p1);
  vn_head_kernel<<<128, TPB, 0, stream>>>(p2);
  vn_final_kernel<<<1, TPB, 0, stream>>>(value, adv, (float*)d_out);
}
